// EncBiSRU_36713380446235
// MI455X (gfx1250) — matrix operand order repaired and checked by reading, not compiled, not run on hardware
//
#include <hip/hip_runtime.h>
#include <hip/hip_bf16.h>
#include <stdint.h>

#define B_ 128
#define L_ 2048
#define HID_ 32
#define NL_ 4
#define EPS_ 1e-5f
#define M_ (L_ * B_)      // 262144 rows
#define NCAT 256          // 2 dirs * 4*HID outputs
#define KDIM 64           // 2*HID input features
#define TPB 8             // A-tiles (of 64 rows) per persistent GEMM block

typedef __attribute__((ext_vector_type(16))) __bf16 v16bf;
typedef __attribute__((ext_vector_type(8)))  float  v8f;

__device__ __forceinline__ unsigned bf16_rne(float x) {
  unsigned u = __builtin_bit_cast(unsigned, x);
  return (u + 0x7FFFu + ((u >> 16) & 1u)) >> 16;
}
__device__ __forceinline__ unsigned pack2(float lo, float hi) {
  return bf16_rne(lo) | (bf16_rne(hi) << 16);
}
__device__ __forceinline__ float sigmoidf_(float x) {
  return 1.0f / (1.0f + __expf(-x));
}

// ---- async global->LDS copy of 16B per lane (falls back to VGPR copy) ----
__device__ __forceinline__ void async_cp16(const void* g, void* l) {
#if __has_builtin(__builtin_amdgcn_global_load_async_to_lds_b128)
  typedef __attribute__((ext_vector_type(4))) int v4i_t;
  typedef __attribute__((address_space(1))) v4i_t* gp_t;  // global
  typedef __attribute__((address_space(3))) v4i_t* sp_t;  // LDS
  // generic->AS casts via integer reinterpret: AS1 shares the 64-bit generic
  // representation; generic LDS addr keeps the LDS byte offset in addr[31:0].
  __builtin_amdgcn_global_load_async_to_lds_b128(
      (gp_t)(unsigned long long)(uintptr_t)g,
      (sp_t)(unsigned)(uintptr_t)l, 0, 0);
#else
  *(uint4*)l = *(const uint4*)g;
#endif
}
__device__ __forceinline__ void async_wait0() {
#if __has_builtin(__builtin_amdgcn_s_wait_asynccnt)
  __builtin_amdgcn_s_wait_asynccnt(0);
#elif __has_builtin(__builtin_amdgcn_global_load_async_to_lds_b128)
  asm volatile("s_wait_asynccnt 0x0" ::: "memory");
#endif
}

// ---- Layer 0: D_IN==1, LN over one feature + K=1 "GEMM" (outer product) ----
__global__ __launch_bounds__(256) void layer0_u(
    const float* __restrict__ x, const float* __restrict__ W0,
    const float* __restrict__ g, const float* __restrict__ bb,
    float* __restrict__ U)
{
  int m = blockIdx.x;           // row = l*B + b
  int n = threadIdx.x;          // 0..255 (dir*128 + gate-col)
  int b = m & (B_ - 1);
  int l = m >> 7;
  float xv = x[(size_t)b * L_ + l];
  float mu = xv;                // mean over D_IN=1
  float xnv = (xv - mu) * rsqrtf(0.0f + EPS_) * g[0] + bb[0];
  U[(size_t)m * NCAT + n] = xnv * W0[n]; // W0 flat (2,1,128) == n
}

// ---- Convert W_rest (3,2,64,128) f32 -> per-layer concat (64,256) bf16 ----
__global__ __launch_bounds__(256) void make_wcat(
    const float* __restrict__ Wr, unsigned short* __restrict__ Wcat)
{
  int tid = blockIdx.x * blockDim.x + threadIdx.x; // < 3*2*64*128 = 49152
  int l1   = tid >> 14;
  int rem  = tid & 16383;
  int d    = rem >> 13;
  int rem2 = rem & 8191;         // i*128 + k
  int i    = rem2 >> 7;
  int k    = rem2 & 127;
  float w = Wr[((size_t)(l1 * 2 + d) << 13) + rem2];
  Wcat[(size_t)l1 * 16384 + i * 256 + d * 128 + k] = (unsigned short)bf16_rne(w);
}

// ---- LayerNorm(64) f32 -> packed bf16; 32 lanes per row, wave reductions ----
__global__ __launch_bounds__(256) void ln_to_bf16(
    const float* __restrict__ h, const float* __restrict__ g,
    const float* __restrict__ bb, unsigned* __restrict__ xn_pk)
{
  int tid  = blockIdx.x * blockDim.x + threadIdx.x;
  int lane = tid & 31;
  int row  = tid >> 5;
  const float2 val = ((const float2*)(h + (size_t)row * KDIM))[lane];
  float s = val.x + val.y;
#pragma unroll
  for (int off = 16; off > 0; off >>= 1) s += __shfl_xor(s, off, 32);
  float mu = s * (1.0f / KDIM);
  float dx = val.x - mu, dy = val.y - mu;
  float vs = dx * dx + dy * dy;
#pragma unroll
  for (int off = 16; off > 0; off >>= 1) vs += __shfl_xor(vs, off, 32);
  float inv = rsqrtf(vs * (1.0f / KDIM) + EPS_);
  float ox = dx * inv * g[2 * lane]     + bb[2 * lane];
  float oy = dy * inv * g[2 * lane + 1] + bb[2 * lane + 1];
  xn_pk[(size_t)row * 32 + lane] = pack2(ox, oy);
}

// ---- GEMM: (M,64)bf16 @ (64,256)bf16 -> (M,256)f32 via v_wmma_f32_16x16x32_bf16
// Persistent block: 256 thr (8 waves); weights LDS-resident for TPB A-tiles;
// A tiles double-buffered via async global->LDS, overlapped with WMMA compute.
__global__ __launch_bounds__(256) void gemm_bf16_wmma(
    const unsigned short* __restrict__ xn,
    const unsigned short* __restrict__ Wcat,
    float* __restrict__ U)
{
  __shared__ unsigned short lA[2][64 * KDIM]; // 2 x 8 KB
  __shared__ unsigned short lB[KDIM * NCAT];  // 32 KB
  const int tid = threadIdx.x;

  { // stage B (weights, 32KB): 8 x 16B per thread, async
    const char* g = (const char*)Wcat;
    char* s = (char*)lB;
#pragma unroll
    for (int i = 0; i < 8; ++i)
      async_cp16(g + (size_t)(tid + 256 * i) * 16, s + (size_t)(tid + 256 * i) * 16);
  }
  { // stage first A tile (8KB): 2 x 16B per thread, async
    const char* g = (const char*)(xn + (size_t)blockIdx.x * TPB * 64 * KDIM);
    char* s = (char*)lA[0];
#pragma unroll
    for (int i = 0; i < 2; ++i)
      async_cp16(g + (size_t)(tid + 256 * i) * 16, s + (size_t)(tid + 256 * i) * 16);
  }
  async_wait0();
  __syncthreads();

  const int lane  = tid & 31;
  const int wave  = tid >> 5;
  const int half  = lane >> 4;     // which 16-lane half
  const int lm    = lane & 15;     // A row / C col within tile
  const int mbase = (wave >> 1) * 16;

  for (int it = 0; it < TPB; ++it) {
    const int m0 = (blockIdx.x * TPB + it) * 64;

    if (it + 1 < TPB) { // kick off next A tile while computing this one
      const char* g = (const char*)(xn + (size_t)(m0 + 64) * KDIM);
      char* s = (char*)lA[(it + 1) & 1];
#pragma unroll
      for (int i = 0; i < 2; ++i)
        async_cp16(g + (size_t)(tid + 256 * i) * 16, s + (size_t)(tid + 256 * i) * 16);
    }

    const unsigned short* A = lA[it & 1];
    // A fragments per ISA 16-bit A 16x32 layout; two K-chunks of 32
    unsigned araw[2][8];
#pragma unroll
    for (int kc = 0; kc < 2; ++kc)
#pragma unroll
      for (int v = 0; v < 8; ++v) {
        int k0 = ((v & 3) << 1) + (half << 3) + ((v >> 2) << 4) + (kc << 5);
        araw[kc][v] = *(const unsigned*)&A[(mbase + lm) * KDIM + k0];
      }
    v16bf afrag[2];
    __builtin_memcpy(&afrag[0], &araw[0][0], 32);
    __builtin_memcpy(&afrag[1], &araw[1][0], 32);

#pragma unroll
    for (int i = 0; i < 8; ++i) {
      const int nt = ((wave & 1) << 3) + i;   // 16 N-tiles split over wave pairs
      const int n0 = nt << 4;
      v8f acc = {};
#pragma unroll
      for (int kc = 0; kc < 2; ++kc) {
        unsigned braw[8];
        // B column n0+lm per lane; K in same table order as A
#pragma unroll
        for (int v = 0; v < 8; ++v) {
          int k0 = ((v & 3) << 1) + (half << 3) + ((v >> 2) << 4) + (kc << 5);
          braw[v] = (unsigned)lB[k0 * NCAT + n0 + lm] |
                    ((unsigned)lB[(k0 + 1) * NCAT + n0 + lm] << 16);
        }
        v16bf bfrag;
        __builtin_memcpy(&bfrag, braw, 32);
        acc = __builtin_amdgcn_wmma_f32_16x16x32_bf16(
            false, afrag[kc], false, bfrag, (short)0, acc, false, false);
      }
      // C/D layout: VGPR r -> M = r + 8*half, N = lm
      float* dst = U + (size_t)(m0 + mbase + (half << 3)) * NCAT + n0 + lm;
#pragma unroll
      for (int r = 0; r < 8; ++r)
        dst[(size_t)r * NCAT] = acc[r];
    }

    async_wait0();     // next tile's transfers done (this wave's)
    __syncthreads();   // ...and everyone else's, before consuming/overwriting
  }
}

// ---- Sequential SRU scan: 8192 chains (dir,b,hid); fwd walks l up, bwd down
__global__ __launch_bounds__(256) void sru_scan(
    const float* __restrict__ U,   // (L,B,2,128)
    const float* __restrict__ v,   // (NL,2,2,32)
    const float* __restrict__ bb,  // (NL,2,2,32)
    int layer,
    float* __restrict__ hout,      // (L,B,64)
    float* __restrict__ cfin)      // (B,64) slice of d_out h_n
{
  int tid = blockIdx.x * blockDim.x + threadIdx.x;  // 0..8191
  int j = tid & 31;
  int b = (tid >> 5) & (B_ - 1);
  int d = tid >> 12;
  const float* vl = v  + (size_t)((layer * 2 + d) * 2) * HID_;
  const float* bl = bb + (size_t)((layer * 2 + d) * 2) * HID_;
  float vf = vl[j], vr = vl[HID_ + j];
  float bf = bl[j], br = bl[HID_ + j];

  const long ustep = (long)B_ * NCAT * (d ? -1 : 1);
  const long hstep = (long)B_ * KDIM * (d ? -1 : 1);
  const float* Urow = U + ((size_t)(d ? (L_ - 1) : 0) * B_ + b) * NCAT + d * 128 + j;
  float*       hrow = hout + ((size_t)(d ? (L_ - 1) : 0) * B_ + b) * KDIM + d * HID_ + j;

  float c = 0.f;
  for (int t = 0; t < L_; ++t) {
    float u0 = Urow[0], u1 = Urow[32], u2 = Urow[64], u3 = Urow[96];
    if (t + 1 < L_) __builtin_prefetch(Urow + ustep, 0, 1);  // global_prefetch_b8
    float f  = sigmoidf_(u1 + vf * c + bf);
    float cn = f * c + (1.f - f) * u0;
    float r  = sigmoidf_(u2 + vr * c + br);
    *hrow = r * cn + (1.f - r) * u3;
    c = cn;
    Urow += ustep; hrow += hstep;
  }
  cfin[b * KDIM + d * HID_ + j] = c;
}

// ---- Final heads: context = h_last[L-1] @ Wd/Wl + biases ----
__global__ void head_kernel(const float* __restrict__ hlast,
                            const float* __restrict__ Wd, const float* __restrict__ bd,
                            const float* __restrict__ Wl, const float* __restrict__ bl,
                            float* __restrict__ out)
{
  int b = threadIdx.x;  // 128 threads
  const float* ctx = hlast + ((size_t)(L_ - 1) * B_ + b) * KDIM;
  float sd = 0.f, sl = 0.f;
  for (int i = 0; i < KDIM; ++i) { float c = ctx[i]; sd += c * Wd[i]; sl += c * Wl[i]; }
  out[b]      = sd + bd[0];
  out[B_ + b] = sl + bl[0];
}

extern "C" void kernel_launch(void* const* d_in, const int* in_sizes, int n_in,
                              void* d_out, int out_size, void* d_ws, size_t ws_size,
                              hipStream_t stream) {
  const float* x    = (const float*)d_in[0];
  const float* W0   = (const float*)d_in[1];
  const float* Wr   = (const float*)d_in[2];
  const float* v    = (const float*)d_in[3];
  const float* b    = (const float*)d_in[4];
  const float* ln0g = (const float*)d_in[5];
  const float* ln0b = (const float*)d_in[6];
  const float* lng  = (const float*)d_in[7];
  const float* lnb  = (const float*)d_in[8];
  const float* Wd   = (const float*)d_in[9];
  const float* bd   = (const float*)d_in[10];
  const float* Wl   = (const float*)d_in[11];
  const float* bl   = (const float*)d_in[12];
  float* out = (float*)d_out;

  // workspace carve-up
  float*    U    = (float*)d_ws;                               // M*256 f32
  float*    hA   = U  + (size_t)M_ * NCAT;                     // M*64 f32
  float*    hB   = hA + (size_t)M_ * KDIM;                     // M*64 f32
  unsigned* xn   = (unsigned*)(hB + (size_t)M_ * KDIM);        // M*32 dwords (bf16x2)
  unsigned short* Wcat = (unsigned short*)(xn + (size_t)M_ * 32); // 3*64*256 bf16

  make_wcat<<<192, 256, 0, stream>>>(Wr, Wcat);

  // layer 0 (degenerate K=1)
  layer0_u<<<M_, 256, 0, stream>>>(x, W0, ln0g, ln0b, U);
  sru_scan<<<32, 256, 0, stream>>>(U, v, b, 0, hA, out + 2 * B_);

  float* hin = hA; float* hout = hB;
  for (int l = 1; l < NL_; ++l) {
    ln_to_bf16<<<M_ / 8, 256, 0, stream>>>(hin, lng + (size_t)(l - 1) * KDIM,
                                           lnb + (size_t)(l - 1) * KDIM, xn);
    gemm_bf16_wmma<<<M_ / (64 * TPB), 256, 0, stream>>>(
        (const unsigned short*)xn, Wcat + (size_t)(l - 1) * KDIM * NCAT, U);
    sru_scan<<<32, 256, 0, stream>>>(U, v, b, l, hout,
                                     out + 2 * B_ + (size_t)l * B_ * KDIM);
    float* t = hin; hin = hout; hout = t;
  }

  head_kernel<<<1, B_, 0, stream>>>(hin, Wd, bd, Wl, bl, out);
}
